// TeacherNet_84189948936790
// MI455X (gfx1250) — matrix operand order repaired and checked by reading, not compiled, not run on hardware
//
#include <hip/hip_runtime.h>
#include <cstdint>

// ---------------------------------------------------------------------------
// 3-layer GAT for MI455X (gfx1250, wave32).
//  - GEMMs via v_wmma_f32_16x16x32_bf16 (f32 accumulate), double-buffered LDS
//    pipeline: global_load_b128 (next tile) overlaps WMMA (current tile);
//    one s_barrier per K-step. f32->bf16 staging uses __builtin_convertvector
//    so ISel emits paired v_cvt_pk_bf16_f32 directly (no half-register repack).
//  - message passing via coalesced wave-per-edge gather + global_atomic_add_f32
//  - segment softmax via int-trick atomic max + atomic add
// Workspace requirement: ~134 MB (3x 20000x512 f32 + edge scratch).
// ---------------------------------------------------------------------------

typedef __attribute__((ext_vector_type(16))) __bf16 v16bf;
typedef __attribute__((ext_vector_type(8)))  float  v8f;
typedef __attribute__((ext_vector_type(4)))  float  v4f;
typedef __attribute__((ext_vector_type(2)))  float  v2f;
typedef __attribute__((ext_vector_type(4)))  __bf16 v4bf;
typedef __attribute__((ext_vector_type(2)))  __bf16 v2bf;

#define NN    20000
#define E0    160000
#define ETOT  (E0 + NN)
#define KIN   512
#define TM    64
#define TN    64
#define TK    32
#define ROWP  40          // padded LDS row (elems): 80 bytes => 16B-aligned rows

union FragAB { v16bf v; uint4 q[2]; };   // 32B: A/B bf16 fragment
union FragC  { v8f   v; float f[8]; };   // 32B: C/D f32 fragment

// vector f32->bf16 converts: single IR fptrunc -> paired v_cvt_pk_bf16_f32
__device__ __forceinline__ void cvt_store4(__bf16* dst, float4 v)
{
    v4f t; t.x = v.x; t.y = v.y; t.z = v.z; t.w = v.w;
    *(v4bf*)dst = __builtin_convertvector(t, v4bf);      // ds_store_b64
}
__device__ __forceinline__ void cvt_store2(__bf16* dst, float a, float b)
{
    v2f t; t.x = a; t.y = b;
    *(v2bf*)dst = __builtin_convertvector(t, v2bf);      // ds_store_b32
}

// ---- staged-tile helpers (128 threads) ------------------------------------
__device__ __forceinline__ void load_tileA(const float* __restrict__ A, int bm,
                                           int M, int K, int k0, int tid,
                                           bool interior, float4 (&ra)[4])
{
    #pragma unroll
    for (int it = 0; it < 4; ++it) {
        int f = it * 128 + tid;               // 0..511 : 64 rows x 8 float4
        int r = f >> 3, c4 = f & 7;
        if (interior || (bm + r) < M)
            ra[it] = *(const float4*)(A + (size_t)(bm + r) * K + k0 + c4 * 4);
        else
            ra[it] = make_float4(0.f, 0.f, 0.f, 0.f);
    }
}
__device__ __forceinline__ void load_tileB(const float* __restrict__ B, int bn,
                                           int Nd, int k0, int tid, float4 (&rb)[4])
{
    #pragma unroll
    for (int it = 0; it < 2; ++it) {
        int f  = it * 128 + tid;              // 0..255 : 16 row-pairs x 16 float4
        int rp = f >> 4, c = (f & 15) * 4;
        const float* p = B + (size_t)(k0 + 2 * rp) * Nd + bn + c;
        rb[it * 2 + 0] = *(const float4*)(p);
        rb[it * 2 + 1] = *(const float4*)(p + Nd);
    }
}
__device__ __forceinline__ void store_tileA(__bf16 (*sA)[ROWP], int tid,
                                            const float4 (&ra)[4])
{
    #pragma unroll
    for (int it = 0; it < 4; ++it) {
        int f = it * 128 + tid;
        int r = f >> 3, c4 = f & 7;
        cvt_store4(&sA[r][c4 * 4], ra[it]);
    }
}
__device__ __forceinline__ void store_tileB(__bf16 (*sBt)[ROWP], int tid,
                                            const float4 (&rb)[4])
{
    #pragma unroll
    for (int it = 0; it < 2; ++it) {
        int f  = it * 128 + tid;
        int rp = f >> 4, c = (f & 15) * 4;
        float4 v0 = rb[it * 2 + 0], v1 = rb[it * 2 + 1];
        // (row 2rp, row 2rp+1) land on adjacent K slots in N-major LDS
        cvt_store2(&sBt[c + 0][2 * rp], v0.x, v1.x);
        cvt_store2(&sBt[c + 1][2 * rp], v0.y, v1.y);
        cvt_store2(&sBt[c + 2][2 * rp], v0.z, v1.z);
        cvt_store2(&sBt[c + 3][2 * rp], v0.w, v1.w);
    }
}

// ---------------------------------------------------------------------------
// GEMM: C[M,Nd] = A[M,K] @ B[K,Nd], f32 in/out, bf16 WMMA compute.
// grid = ((M+63)/64, Nd/64), block = 128 (4 waves; wave w owns rows w*16..+15)
// K % 32 == 0, Nd % 64 == 0 (true for all layers: 512/384/64).
// ---------------------------------------------------------------------------
__global__ __launch_bounds__(128)
void gemm_wmma_bf16(const float* __restrict__ A, const float* __restrict__ B,
                    float* __restrict__ C, int M, int K, int Nd)
{
    __shared__ __bf16 sA [2][TM][ROWP];   // A tiles, row-major (M x K)
    __shared__ __bf16 sBt[2][TN][ROWP];   // B tiles, N-major   (N x K)

    const int tid  = threadIdx.x;
    const int wave = tid >> 5;
    const int lane = tid & 31;
    const int bm   = blockIdx.x * TM;
    const int bn   = blockIdx.y * TN;
    const int half = lane >> 4;        // ISA 7.12.2: lanes 16-31 hold +8 / +16 K-halves
    const int l15  = lane & 15;
    const bool interior = (bm + TM) <= M;   // uniform: only last M-block is edge

    FragC acc[4];
    #pragma unroll
    for (int i = 0; i < 4; ++i)
        #pragma unroll
        for (int v = 0; v < 8; ++v) acc[i].f[v] = 0.f;

    float4 ra[4], rb[4];
    load_tileA(A, bm, M, K, 0, tid, interior, ra);
    load_tileB(B, bn, Nd, 0, tid, rb);
    store_tileA(sA[0], tid, ra);
    store_tileB(sBt[0], tid, rb);
    __syncthreads();

    const int nk = K / TK;
    for (int ki = 0; ki < nk; ++ki) {
        const int p = ki & 1;
        const bool more = (ki + 1) < nk;
        if (more) {                                   // overlap: next tile in flight
            load_tileA(A, bm, M, K, (ki + 1) * TK, tid, interior, ra);
            load_tileB(B, bn, Nd, (ki + 1) * TK, tid, rb);
        }

        // A fragment (16x32 bf16): lane half selects K {0..7,16..23} vs {8..15,24..31}
        FragAB af;
        {
            const __bf16* arow = &sA[p][wave * 16 + l15][0];
            af.q[0] = *(const uint4*)(arow + half * 8);        // K 0..7   (+8 if half)
            af.q[1] = *(const uint4*)(arow + 16 + half * 8);   // K 16..23 (+8 if half)
        }
        #pragma unroll
        for (int nb = 0; nb < 4; ++nb) {
            // B fragment (32x16 bf16): n = lane&15, k = {0..7,16..23} (+8 if half), matching A
            FragAB bfr;
            const __bf16* brow = &sBt[p][nb * 16 + l15][0];
            bfr.q[0] = *(const uint4*)(brow + half * 8);
            bfr.q[1] = *(const uint4*)(brow + 16 + half * 8);
            acc[nb].v = __builtin_amdgcn_wmma_f32_16x16x32_bf16(
                false, af.v, false, bfr.v, (short)0, acc[nb].v, false, false);
        }

        if (more) {
            store_tileA(sA[p ^ 1], tid, ra);
            store_tileB(sBt[p ^ 1], tid, rb);
            __syncthreads();   // one barrier per K-step (double-buffered)
        }
    }

    // C/D layout: m = vgpr + (lane/16)*8, n = lane&15  -> coalesced stores
    const int mhi = (lane >> 4) * 8;
    #pragma unroll
    for (int nb = 0; nb < 4; ++nb)
        #pragma unroll
        for (int v = 0; v < 8; ++v) {
            int gm = bm + wave * 16 + mhi + v;
            if (gm < M) C[(size_t)gm * Nd + (bn + nb * 16 + l15)] = acc[nb].f[v];
        }
}

// ---------------------------------------------------------------------------
// Edge list with self-loops (PyG add_self_loops=True)
// ---------------------------------------------------------------------------
__global__ __launch_bounds__(256)
void build_edges(const int* __restrict__ ei, int* __restrict__ s, int* __restrict__ d)
{
    int i = blockIdx.x * 256 + threadIdx.x;
    if (i >= ETOT) return;
    if (i < E0) { s[i] = ei[i]; d[i] = ei[E0 + i]; }
    else        { s[i] = i - E0; d[i] = i - E0; }
}

// ---------------------------------------------------------------------------
// Attention coefficients: one wave32 per (node, head). Lanes do float4 partial
// dots over the contiguous 512B/256B row, then a 5-step shfl_xor butterfly.
// Also resets the softmax max/denom buffers.
// ---------------------------------------------------------------------------
__global__ __launch_bounds__(256)
void attn_coef(const float* __restrict__ hW, const float* __restrict__ attS,
               const float* __restrict__ attD, float* __restrict__ aS,
               float* __restrict__ aD, float* __restrict__ mB,
               float* __restrict__ dB, int H, int C)
{
    int gid  = blockIdx.x * 256 + threadIdx.x;
    int w    = gid >> 5;              // (n*H + h)
    int lane = gid & 31;
    if (w >= NN * H) return;
    int n = w / H, h = w - n * H;

    const float4* base = (const float4*)(hW + (size_t)n * H * C + (size_t)h * C);
    const float4* ws   = (const float4*)(attS + h * C);
    const float4* wd   = (const float4*)(attD + h * C);

    float ss = 0.f, sd = 0.f;
    for (int c4 = lane; c4 < (C >> 2); c4 += 32) {
        float4 v = base[c4], s4 = ws[c4], d4 = wd[c4];
        ss += v.x * s4.x + v.y * s4.y + v.z * s4.z + v.w * s4.w;
        sd += v.x * d4.x + v.y * d4.y + v.z * d4.z + v.w * d4.w;
    }
    #pragma unroll
    for (int o = 16; o > 0; o >>= 1) {
        ss += __shfl_xor(ss, o, 32);
        sd += __shfl_xor(sd, o, 32);
    }
    if (lane == 0) {
        aS[w] = ss; aD[w] = sd;
        mB[w] = -3.0e38f; dB[w] = 0.f;
    }
}

// float atomic-max via sign-aware int/uint atomics (ordered encoding trick)
__device__ __forceinline__ void atomicMaxF(float* addr, float val)
{
    if (val >= 0.f) atomicMax((int*)addr, __float_as_int(val));
    else            atomicMin((unsigned int*)addr, __float_as_uint(val));
}

__global__ __launch_bounds__(256)
void edge_max(const int* __restrict__ s, const int* __restrict__ d,
              const float* __restrict__ aS, const float* __restrict__ aD,
              float* __restrict__ eV, float* __restrict__ mB, int H)
{
    int idx = blockIdx.x * 256 + threadIdx.x;
    if (idx >= ETOT * H) return;
    int e = idx / H, h = idx - e * H;
    float v = aS[s[e] * H + h] + aD[d[e] * H + h];
    v = (v > 0.f) ? v : 0.2f * v;           // leaky_relu, slope 0.2
    eV[idx] = v;
    atomicMaxF(&mB[d[e] * H + h], v);
}

__global__ __launch_bounds__(256)
void edge_exp(const int* __restrict__ d, float* __restrict__ eV,
              const float* __restrict__ mB, float* __restrict__ dB, int H)
{
    int idx = blockIdx.x * 256 + threadIdx.x;
    if (idx >= ETOT * H) return;
    int e = idx / H, h = idx - e * H;
    float ex = __expf(eV[idx] - mB[d[e] * H + h]);
    eV[idx] = ex;
    atomicAdd(&dB[d[e] * H + h], ex);
}

__global__ __launch_bounds__(256)
void edge_alpha(const int* __restrict__ d, float* __restrict__ eV,
                const float* __restrict__ dB, int H)
{
    int idx = blockIdx.x * 256 + threadIdx.x;
    if (idx >= ETOT * H) return;
    int e = idx / H, h = idx - e * H;
    eV[idx] = eV[idx] / (dB[d[e] * H + h] + 1e-16f);
}

// ---------------------------------------------------------------------------
// Message scatter: one wave per edge; float4 gathers from hW[src] (L2-resident
// at 41MB << 192MB L2), scalar global_atomic_add_f32 into accumulator.
// ---------------------------------------------------------------------------
__global__ __launch_bounds__(256)
void edge_scatter(const int* __restrict__ s, const int* __restrict__ d,
                  const float* __restrict__ hW, const float* __restrict__ alpha,
                  float* __restrict__ gat, int H, int C)
{
    int gid  = blockIdx.x * 256 + threadIdx.x;
    int e    = gid >> 5;
    int lane = gid & 31;
    if (e >= ETOT) return;

    const int HC = H * C;
    const int C4 = C >> 2;                       // float4s per head
    const int se = s[e], de = d[e];
    const float4* hsrc = (const float4*)(hW + (size_t)se * HC);
    float*        outp = gat + (size_t)de * HC;
    const float*  al   = alpha + (size_t)e * H;

    if (e + 8 < ETOT && lane == 0)               // warm L2/L0 for an upcoming edge
        __builtin_prefetch(hW + (size_t)s[e + 8] * HC, 0, 1);

    for (int c4 = lane; c4 < (HC >> 2); c4 += 32) {
        float  a = al[c4 / C4];
        float4 v = hsrc[c4];
        int    c = c4 << 2;
        atomicAdd(&outp[c + 0], v.x * a);
        atomicAdd(&outp[c + 1], v.y * a);
        atomicAdd(&outp[c + 2], v.z * a);
        atomicAdd(&outp[c + 3], v.w * a);
    }
}

// ---------------------------------------------------------------------------
// Epilogues
// ---------------------------------------------------------------------------
__global__ __launch_bounds__(256)
void epilogue_elu(const float* __restrict__ gat, const float* __restrict__ b,
                  const float* __restrict__ lb, float* __restrict__ out,
                  int total, int F)
{
    int idx = blockIdx.x * 256 + threadIdx.x;
    if (idx >= total) return;
    int f = idx & (F - 1);                       // F is a power of two (512)
    float v = gat[idx] + b[f] + lb[f];
    out[idx] = (v > 0.f) ? v : (__expf(v) - 1.f);
}

__global__ __launch_bounds__(256)
void epilogue_mean(const float* __restrict__ gat, const float* __restrict__ lin,
                   const float* __restrict__ b, const float* __restrict__ lb,
                   float* __restrict__ out, int H, int C)
{
    int idx = blockIdx.x * 256 + threadIdx.x;
    if (idx >= NN * C) return;
    int n = idx / C, c = idx - n * C;
    const float* g = gat + (size_t)n * H * C + c;
    float sum = 0.f;
    for (int h = 0; h < H; ++h) sum += g[h * C];
    out[idx] = sum * (1.f / (float)H) + b[c] + lin[idx] + lb[c];
}

__global__ __launch_bounds__(256)
void zero_f32(float* __restrict__ p, int n)
{
    int idx = blockIdx.x * 256 + threadIdx.x;
    if (idx < n) p[idx] = 0.f;
}

// ---------------------------------------------------------------------------
// Host-side orchestration (stream-ordered; graph-capture safe)
// ---------------------------------------------------------------------------
static void launch_gemm(const float* A, const float* B, float* C,
                        int M, int K, int Nd, hipStream_t st)
{
    dim3 g((M + TM - 1) / TM, Nd / TN);
    gemm_wmma_bf16<<<g, 128, 0, st>>>(A, B, C, M, K, Nd);
}

static void gat_layer(const float* hW, const float* attS, const float* attD,
                      float* gatAcc, float* aS, float* aD, float* mB, float* dB,
                      float* eV, const int* srcI, const int* dstI,
                      int H, int C, hipStream_t st)
{
    int nhw = NN * H * 32;   // one wave per (node, head)
    attn_coef<<<(nhw + 255) / 256, 256, 0, st>>>(hW, attS, attD, aS, aD, mB, dB, H, C);
    int eh = ETOT * H;
    edge_max  <<<(eh + 255) / 256, 256, 0, st>>>(srcI, dstI, aS, aD, eV, mB, H);
    edge_exp  <<<(eh + 255) / 256, 256, 0, st>>>(dstI, eV, mB, dB, H);
    edge_alpha<<<(eh + 255) / 256, 256, 0, st>>>(dstI, eV, dB, H);
    edge_scatter<<<(ETOT * 32 + 255) / 256, 256, 0, st>>>(srcI, dstI, hW, eV, gatAcc, H, C);
}

extern "C" void kernel_launch(void* const* d_in, const int* in_sizes, int n_in,
                              void* d_out, int out_size, void* d_ws, size_t ws_size,
                              hipStream_t stream)
{
    (void)in_sizes; (void)n_in; (void)out_size; (void)ws_size;

    const float* x   = (const float*)d_in[0];
    const int*   ei  = (const int*)  d_in[1];
    const float* W1  = (const float*)d_in[2];
    const float* as1 = (const float*)d_in[3];
    const float* ad1 = (const float*)d_in[4];
    const float* b1  = (const float*)d_in[5];
    const float* lW1 = (const float*)d_in[6];
    const float* lb1 = (const float*)d_in[7];
    const float* W2  = (const float*)d_in[8];
    const float* as2 = (const float*)d_in[9];
    const float* ad2 = (const float*)d_in[10];
    const float* b2  = (const float*)d_in[11];
    const float* lW2 = (const float*)d_in[12];
    const float* lb2 = (const float*)d_in[13];
    const float* W3  = (const float*)d_in[14];
    const float* as3 = (const float*)d_in[15];
    const float* ad3 = (const float*)d_in[16];
    const float* b3  = (const float*)d_in[17];
    const float* lW3 = (const float*)d_in[18];
    const float* lb3 = (const float*)d_in[19];
    float* out = (float*)d_out;

    // -------- workspace carve (f32 units) --------
    float* bufW = (float*)d_ws;                    // transformed features [N,512]
    float* bufG = bufW + (size_t)NN * 512;         // GAT accumulator     [N,512]
    float* bufA = bufG + (size_t)NN * 512;         // layer activations   [N,512]
    float* lin3 = bufA + (size_t)NN * 512;         // layer3 linear skip  [N,64]
    float* aS   = lin3 + (size_t)NN * 64;          // [N,6]
    float* aD   = aS   + (size_t)NN * 6;
    float* mB   = aD   + (size_t)NN * 6;
    float* dB   = mB   + (size_t)NN * 6;
    float* eV   = dB   + (size_t)NN * 6;           // [ETOT,6]
    int*   srcI = (int*)(eV + (size_t)ETOT * 6);
    int*   dstI = srcI + ETOT;

    build_edges<<<(ETOT + 255) / 256, 256, 0, stream>>>(ei, srcI, dstI);

    // ---------------- Layer 1 (H=4, C=128) ----------------
    launch_gemm(x, lW1, bufG, NN, KIN, 512, stream);   // linear skip seeds accumulator
    launch_gemm(x, W1,  bufW, NN, KIN, 512, stream);
    gat_layer(bufW, as1, ad1, bufG, aS, aD, mB, dB, eV, srcI, dstI, 4, 128, stream);
    epilogue_elu<<<(NN * 512 + 255) / 256, 256, 0, stream>>>(bufG, b1, lb1, bufA, NN * 512, 512);

    // ---------------- Layer 2 (H=4, C=128) ----------------
    launch_gemm(bufA, lW2, bufG, NN, 512, 512, stream);
    launch_gemm(bufA, W2,  bufW, NN, 512, 512, stream);
    gat_layer(bufW, as2, ad2, bufG, aS, aD, mB, dB, eV, srcI, dstI, 4, 128, stream);
    epilogue_elu<<<(NN * 512 + 255) / 256, 256, 0, stream>>>(bufG, b2, lb2, bufA, NN * 512, 512);

    // ---------------- Layer 3 (H=6, C=64, mean over heads) ----------------
    launch_gemm(bufA, lW3, lin3, NN, 512, 64, stream);
    launch_gemm(bufA, W3,  bufW, NN, 512, 384, stream);
    zero_f32<<<(NN * 384 + 255) / 256, 256, 0, stream>>>(bufG, NN * 384);
    gat_layer(bufW, as3, ad3, bufG, aS, aD, mB, dB, eV, srcI, dstI, 6, 64, stream);
    epilogue_mean<<<(NN * 64 + 255) / 256, 256, 0, stream>>>(bufG, lin3, b3, lb3, out, 6, 64);
}
